// RNet2_13417477833447
// MI455X (gfx1250) — matrix operand order repaired and checked by reading, not compiled, not run on hardware
//
#include <hip/hip_runtime.h>
#include <hip/hip_bf16.h>

#define NBLOCKS  24
#define NTHREADS 256
#define NWAVES   (NBLOCKS * NTHREADS / 32)   // 192 waves

typedef __bf16 bf16t;
typedef __attribute__((ext_vector_type(8)))  __bf16 v8bf;
typedef __attribute__((ext_vector_type(16))) __bf16 v16bf;
typedef __attribute__((ext_vector_type(8)))  float  v8f;

// Explicit global-address-space pointee types: loads through these must
// select global_load_* (LOADcnt only) instead of flat_* (LOADcnt+DScnt).
#define AS1 __attribute__((address_space(1)))
typedef AS1 const v8bf  gv8bf;
typedef AS1 const v16bf gv16bf;

// GRU index: 0=m, 1=h1, 2=h2, 3=k1, 4=k2, 5=f1, 6=f2
// xin[g] is the zero-padded bf16 concatenated input vector for GRU g.
struct Params {
  const float* x;      // (T,1,11) f32
  float*       out;    // (T,1,6)  f32
  int          T;
  const bf16t* wih[7]; // (3H, Kpad) bf16, row-major, zero-padded K
  const bf16t* whh[7]; // (3H, H)    bf16
  const float* bih[7];
  const float* bhh[7];
  const bf16t* wmout;  // (1024,1024) bf16
  const float* bmout;
  const bf16t* wlin[10]; // 512x512 bf16, mid order: h1_dn,h1_up,h2_dn,h2_up,k1_dn,k1_up,k2_dn,k2_up,f1_up,f2_up
  const float* blin[10];
  const float* aw[6];  // action weights f32 (1,512): h1,k1,f1,h2,k2,f2
  const float* ab[6];
  float* stf[7];       // [2][H] f32 states (ping-pong by t&1)
  bf16t* stb[7];       // [2][H] bf16 states
  bf16t* xin[7];       // lens {1056,1056,1056,1056,1056,544,544}
  int*   sync;
};

// ---------------- WMMA helpers (CDNA5 16x16x32 bf16) ----------------

__device__ __forceinline__ v8f wmma_bf16(v16bf a, v16bf b, v8f c) {
  // D = A(16x32 bf16) x B(32x16 bf16) + C(16x16 f32)
  return __builtin_amdgcn_wmma_f32_16x16x32_bf16(false, a, false, b, (short)0, c,
                                                 false, false);
}

// A-layout: lane L (L<16): row m0+L, K = kb+0..15
//           lane L+16   : row m0+L, K = kb+16..31
// => per-lane: two adjacent 8xbf16 (16B) chunks (16 contiguous K).
__device__ __forceinline__ v16bf load_a16(const bf16t* p) {
  v8bf lo = *(gv8bf*)p;
  v8bf hi = *(gv8bf*)(p + 8);
  return __builtin_shufflevector(lo, hi, 0, 1, 2, 3, 4, 5, 6, 7,
                                 8, 9, 10, 11, 12, 13, 14, 15);
}

__device__ __forceinline__ v16bf load_b16(const bf16t* p) {
  return *(gv16bf*)p;
}

// B-layout (all 16 columns = x): lanes 0-15 load x[kb..kb+15], lanes 16-31
// load x[kb+16..kb+31]; every output column then equals W·x.
// Two independent accumulator chains (K step 64) so the scheduler can issue
// back-to-back WMMAs without accumulator copies / hazard NOPs.
template <int KP>
__device__ __forceinline__ v8f dot16(const bf16t* __restrict__ W,
                                     const bf16t* __restrict__ xin,
                                     int m0, int lane) {
  v8f a0 = {}, a1 = {};
  const int offA = (lane & 16) ? 16 : 0;
  const int offB = (lane & 16) ? 16 : 0;
  const bf16t* wrow = W + (size_t)(m0 + (lane & 15)) * KP + offA;
  const bf16t* bp = xin + offB;
  for (int kb = 0; kb + 64 <= KP; kb += 64) {
    v16bf b0 = load_b16(bp + kb);
    v16bf b1 = load_b16(bp + kb + 32);
    a0 = wmma_bf16(load_a16(wrow + kb), b0, a0);
    a1 = wmma_bf16(load_a16(wrow + kb + 32), b1, a1);
  }
  if constexpr ((KP % 64) != 0) {
    constexpr int kb = KP - 32;
    v16bf b = load_b16(bp + kb);
    a0 = wmma_bf16(load_a16(wrow + kb), b, a0);
  }
  return a0 + a1;
}

// Fused GRU tile: rows [m0, m0+16) of h' = (1-z)*n + z*h.
// One wave computes all six gate GEMV tiles; per k-chunk the B vector is
// shared by the three gates, and each gate keeps two accumulator chains.
template <int KIN, int H>
__device__ void gru_tile(const bf16t* __restrict__ Wih,
                         const bf16t* __restrict__ xin,
                         const bf16t* __restrict__ Whh,
                         const bf16t* __restrict__ hprev_bf,
                         const float* __restrict__ bih,
                         const float* __restrict__ bhh,
                         const float* __restrict__ hprev_f,
                         float* __restrict__ hcur_f,
                         bf16t* __restrict__ hcur_bf,
                         int m0, int lane) {
  v8f air0 = {}, aiz0 = {}, ain0 = {}, air1 = {}, aiz1 = {}, ain1 = {};
  const int offA = (lane & 16) ? 16 : 0;
  const int offB = (lane & 16) ? 16 : 0;
  const int rowA = m0 + (lane & 15);
  {
    const bf16t* wr = Wih + (size_t)rowA * KIN + offA;
    const bf16t* wz = wr + (size_t)H * KIN;
    const bf16t* wn = wz + (size_t)H * KIN;
    const bf16t* bx = xin + offB;
    for (int kb = 0; kb + 64 <= KIN; kb += 64) {
      v16bf b0 = load_b16(bx + kb);
      v16bf b1 = load_b16(bx + kb + 32);
      air0 = wmma_bf16(load_a16(wr + kb), b0, air0);
      aiz0 = wmma_bf16(load_a16(wz + kb), b0, aiz0);
      ain0 = wmma_bf16(load_a16(wn + kb), b0, ain0);
      air1 = wmma_bf16(load_a16(wr + kb + 32), b1, air1);
      aiz1 = wmma_bf16(load_a16(wz + kb + 32), b1, aiz1);
      ain1 = wmma_bf16(load_a16(wn + kb + 32), b1, ain1);
    }
    if constexpr ((KIN % 64) != 0) {
      constexpr int kb = KIN - 32;
      v16bf b = load_b16(bx + kb);
      air0 = wmma_bf16(load_a16(wr + kb), b, air0);
      aiz0 = wmma_bf16(load_a16(wz + kb), b, aiz0);
      ain0 = wmma_bf16(load_a16(wn + kb), b, ain0);
    }
  }
  v8f ahr0 = {}, ahz0 = {}, ahn0 = {}, ahr1 = {}, ahz1 = {}, ahn1 = {};
  {
    const bf16t* ur = Whh + (size_t)rowA * H + offA;
    const bf16t* uz = ur + (size_t)H * H;
    const bf16t* un = uz + (size_t)H * H;
    const bf16t* bh = hprev_bf + offB;
    for (int kb = 0; kb + 64 <= H; kb += 64) {
      v16bf b0 = load_b16(bh + kb);
      v16bf b1 = load_b16(bh + kb + 32);
      ahr0 = wmma_bf16(load_a16(ur + kb), b0, ahr0);
      ahz0 = wmma_bf16(load_a16(uz + kb), b0, ahz0);
      ahn0 = wmma_bf16(load_a16(un + kb), b0, ahn0);
      ahr1 = wmma_bf16(load_a16(ur + kb + 32), b1, ahr1);
      ahz1 = wmma_bf16(load_a16(uz + kb + 32), b1, ahz1);
      ahn1 = wmma_bf16(load_a16(un + kb + 32), b1, ahn1);
    }
  }
  // D layout: lane 0 holds rows m0..m0+7 (col 0), lane 16 holds m0+8..m0+15.
  if ((lane & 15) == 0) {
    const int half = lane >> 4;
#pragma unroll
    for (int i = 0; i < 8; ++i) {
      int r = m0 + half * 8 + i;
      float vir = air0[i] + air1[i] + bih[r];
      float viz = aiz0[i] + aiz1[i] + bih[H + r];
      float vin = ain0[i] + ain1[i] + bih[2 * H + r];
      float vhr = ahr0[i] + ahr1[i] + bhh[r];
      float vhz = ahz0[i] + ahz1[i] + bhh[H + r];
      float vhn = ahn0[i] + ahn1[i] + bhh[2 * H + r];
      float rg = 1.0f / (1.0f + __expf(-(vir + vhr)));
      float zg = 1.0f / (1.0f + __expf(-(viz + vhz)));
      float ng = tanhf(vin + rg * vhn);
      float hv = (1.0f - zg) * ng + zg * hprev_f[r];
      hcur_f[r]  = hv;
      hcur_bf[r] = (bf16t)hv;
    }
  }
}

template <int KIN, int H>
__device__ __forceinline__ void run_gru(const Params& p, int g, int m0,
                                        int cur, int prev, int lane) {
  gru_tile<KIN, H>(p.wih[g], p.xin[g], p.whh[g],
                   p.stb[g] + prev * H, p.bih[g], p.bhh[g],
                   p.stf[g] + prev * H, p.stf[g] + cur * H, p.stb[g] + cur * H,
                   m0, lane);
}

// ---------------- grid barrier (counting, non-resetting) ----------------

__device__ __forceinline__ void gbar(int* cnt, int& target) {
  target += NBLOCKS;
  __threadfence();
  __syncthreads();
  if (threadIdx.x == 0) {
    __hip_atomic_fetch_add(cnt, 1, __ATOMIC_ACQ_REL, __HIP_MEMORY_SCOPE_AGENT);
    while (__hip_atomic_load(cnt, __ATOMIC_ACQUIRE, __HIP_MEMORY_SCOPE_AGENT) < target)
      __builtin_amdgcn_s_sleep(2);
  }
  __syncthreads();
  __threadfence();
}

// Scalar observation slots of next step's concatenated inputs.
__device__ __forceinline__ void write_x_scalars(const Params& p, const float* xt) {
  p.xin[0][0] = (bf16t)xt[0];  p.xin[0][1] = (bf16t)xt[1];   // m_obs
  p.xin[0][2] = (bf16t)xt[8];  p.xin[0][3] = (bf16t)xt[9];
  p.xin[0][4] = (bf16t)xt[10];
  p.xin[1][0] = (bf16t)xt[2];  p.xin[2][0] = (bf16t)xt[5];   // h1, h2
  p.xin[3][0] = (bf16t)xt[3];  p.xin[4][0] = (bf16t)xt[6];   // k1, k2
  p.xin[5][0] = (bf16t)xt[4];  p.xin[6][0] = (bf16t)xt[7];   // f1, f2
}

// ---------------- kernels ----------------

__global__ void cvt_kernel(const float* __restrict__ src, bf16t* __restrict__ dst,
                           int dout, int din, int Kp) {
  const long n = (long)dout * Kp;
  for (long i = (long)blockIdx.x * blockDim.x + threadIdx.x; i < n;
       i += (long)gridDim.x * blockDim.x) {
    long r = i / Kp;
    int  k = (int)(i - r * Kp);
    float v = (k < din) ? src[r * din + k] : 0.0f;
    dst[i] = (bf16t)v;
  }
}

__global__ void init_kernel(Params p) {
  const long gt = (long)blockIdx.x * blockDim.x + threadIdx.x;
  const long NT = (long)gridDim.x * blockDim.x;
  for (int g = 0; g < 7; ++g) {
    const int H2 = 2 * ((g == 0) ? 1024 : 512);
    for (long i = gt; i < H2; i += NT) {
      p.stf[g][i] = 0.0f;
      p.stb[g][i] = (bf16t)0.0f;
    }
  }
  const int xl[7] = {1056, 1056, 1056, 1056, 1056, 544, 544};
  for (int g = 0; g < 7; ++g)
    for (long i = gt; i < xl[g]; i += NT) p.xin[g][i] = (bf16t)0.0f;
  if (gt == 0) {
    p.sync[0] = 0;
    write_x_scalars(p, p.x);  // t = 0 observations
  }
}

__global__ __launch_bounds__(NTHREADS) void rnet_kernel(Params p) {
  const int lane = threadIdx.x & 31;
  // Wave index is uniform across the wave: pin it to an SGPR so all phase
  // dispatch is scalar control flow (EXEC stays all-1s around every WMMA).
  const int wgid =
      __builtin_amdgcn_readfirstlane((blockIdx.x * NTHREADS + threadIdx.x) >> 5);
  int bar = 0;
  for (int t = 0; t < p.T; ++t) {
    const int cur = t & 1;
    const int prev = cur ^ 1;

    // ---- P1: m_rnn (64 tiles) + k_rnn1/2 + f_rnn1/2 (32 tiles each) ----
    if (wgid < 64) {
      run_gru<1056, 1024>(p, 0, wgid * 16, cur, prev, lane);
    } else if (wgid < 128) {
      const int g = 3 + ((wgid >> 5) & 1);            // 3 or 4
      run_gru<1056, 512>(p, g, (wgid & 31) * 16, cur, prev, lane);
    } else {
      const int g = 5 + ((wgid >> 5) & 1);            // 5 or 6
      run_gru<544, 512>(p, g, (wgid & 31) * 16, cur, prev, lane);
    }
    gbar(p.sync, bar);

    // ---- P2: out_m = m_out(m_s); written straight into xin_h1/xin_h2 ----
    if (wgid < 64) {
      const int m0 = wgid * 16;
      v8f acc = dot16<1024>(p.wmout, p.stb[0] + cur * 1024, m0, lane);
      if ((lane & 15) == 0) {
        const int half = lane >> 4;
#pragma unroll
        for (int i = 0; i < 8; ++i) {
          int r = m0 + half * 8 + i;
          bf16t yb = (bf16t)(acc[i] + p.bmout[r]);
          if (r < 512) p.xin[1][1 + r] = yb;
          else         p.xin[2][1 + (r - 512)] = yb;
        }
      }
    }
    gbar(p.sync, bar);

    // ---- P3: h_rnn1 / h_rnn2 (32 tiles each) ----
    if (wgid < 64) {
      const int g = 1 + (wgid >> 5);
      run_gru<1056, 512>(p, g, (wgid & 31) * 16, cur, prev, lane);
    }
    gbar(p.sync, bar);

    // ---- P4: 10 x (512x512) up/down linears -> next-step xin slots,
    //          6 action dots -> d_out, + next-step x scalars ----
    for (int u = wgid; u < 327; u += NWAVES) {
      if (u < 320) {
        const int mid = u >> 5;
        const int m0 = (u & 31) * 16;
        int sidx; bf16t* dst;
        switch (mid) {
          case 0: sidx = 1; dst = p.xin[3] + 1;   break;  // h1_down -> k1 in
          case 1: sidx = 1; dst = p.xin[0] + 5;   break;  // h1_up   -> m in
          case 2: sidx = 2; dst = p.xin[4] + 1;   break;  // h2_down -> k2 in
          case 3: sidx = 2; dst = p.xin[0] + 517; break;  // h2_up   -> m in
          case 4: sidx = 3; dst = p.xin[5] + 1;   break;  // k1_down -> f1 in
          case 5: sidx = 3; dst = p.xin[1] + 513; break;  // k1_up   -> h1 in
          case 6: sidx = 4; dst = p.xin[6] + 1;   break;  // k2_down -> f2 in
          case 7: sidx = 4; dst = p.xin[2] + 513; break;  // k2_up   -> h2 in
          case 8: sidx = 5; dst = p.xin[3] + 513; break;  // f1_up   -> k1 in
          default: sidx = 6; dst = p.xin[4] + 513; break; // f2_up   -> k2 in
        }
        v8f acc = dot16<512>(p.wlin[mid], p.stb[sidx] + cur * 512, m0, lane);
        const float* bb = p.blin[mid];
        if ((lane & 15) == 0) {
          const int half = lane >> 4;
#pragma unroll
          for (int i = 0; i < 8; ++i) {
            int r = m0 + half * 8 + i;
            dst[r] = (bf16t)(acc[i] + bb[r]);
          }
        }
      } else if (u < 326) {
        const int aid = u - 320;  // order: h1, k1, f1, h2, k2, f2
        const int smap[6] = {1, 3, 5, 2, 4, 6};
        const float* w = p.aw[aid];
        const float* s = p.stf[smap[aid]] + cur * 512;
        float acc = 0.0f;
        for (int i = lane; i < 512; i += 32) acc += w[i] * s[i];
#pragma unroll
        for (int off = 16; off > 0; off >>= 1) acc += __shfl_down(acc, off, 32);
        if (lane == 0) p.out[t * 6 + aid] = acc + p.ab[aid][0];
      } else {
        if (lane == 0 && (t + 1) < p.T)
          write_x_scalars(p, p.x + (size_t)(t + 1) * 11);
      }
    }
    gbar(p.sync, bar);
  }
}

// ---------------- host launcher ----------------

extern "C" void kernel_launch(void* const* d_in, const int* in_sizes, int n_in,
                              void* d_out, int out_size, void* d_ws, size_t ws_size,
                              hipStream_t stream) {
  (void)n_in; (void)out_size; (void)ws_size;
  // d_in flat order (setup_inputs insertion order): x, then params in _SPECS
  // order, gru = {wih, whh, bih, bhh}, lin = {w, b}.
  static const int IW[7]    = {1, 7, 11, 27, 31, 47, 51};   // gru wih indices
  static const int HH[7]    = {1024, 512, 512, 512, 512, 512, 512};
  static const int DIN[7]   = {1029, 1025, 1025, 1025, 1025, 513, 513};
  static const int KIN[7]   = {1056, 1056, 1056, 1056, 1056, 544, 544};
  static const int ILIN[10] = {19, 23, 21, 25, 39, 43, 41, 45, 59, 61};
  static const int IACT[6]  = {15, 35, 55, 17, 37, 57};

  Params p;
  p.x = (const float*)d_in[0];
  p.out = (float*)d_out;
  p.T = in_sizes[0] / 11;

  char* ws = (char*)d_ws;
  size_t off = 0;
  auto alloc = [&](size_t bytes) -> char* {
    off = (off + 255) & ~(size_t)255;
    char* r = ws + off;
    off += bytes;
    return r;
  };

  bf16t* wih[7]; bf16t* whh[7];
  for (int g = 0; g < 7; ++g) {
    wih[g] = (bf16t*)alloc((size_t)3 * HH[g] * KIN[g] * 2);
    whh[g] = (bf16t*)alloc((size_t)3 * HH[g] * HH[g] * 2);
  }
  bf16t* wmout = (bf16t*)alloc((size_t)1024 * 1024 * 2);
  bf16t* wlin[10];
  for (int m = 0; m < 10; ++m) wlin[m] = (bf16t*)alloc((size_t)512 * 512 * 2);
  for (int g = 0; g < 7; ++g) {
    p.stf[g] = (float*)alloc((size_t)2 * HH[g] * sizeof(float));
    p.stb[g] = (bf16t*)alloc((size_t)2 * HH[g] * 2);
  }
  static const int XL[7] = {1056, 1056, 1056, 1056, 1056, 544, 544};
  for (int g = 0; g < 7; ++g) p.xin[g] = (bf16t*)alloc((size_t)XL[g] * 2);
  p.sync = (int*)alloc(64);

  for (int g = 0; g < 7; ++g) {
    p.wih[g] = wih[g];
    p.whh[g] = whh[g];
    p.bih[g] = (const float*)d_in[IW[g] + 2];
    p.bhh[g] = (const float*)d_in[IW[g] + 3];
  }
  p.wmout = wmout;
  p.bmout = (const float*)d_in[6];
  for (int m = 0; m < 10; ++m) {
    p.wlin[m] = wlin[m];
    p.blin[m] = (const float*)d_in[ILIN[m] + 1];
  }
  for (int a = 0; a < 6; ++a) {
    p.aw[a] = (const float*)d_in[IACT[a]];
    p.ab[a] = (const float*)d_in[IACT[a] + 1];
  }

  // One-time (per launch) f32 -> padded bf16 weight conversion.
  for (int g = 0; g < 7; ++g) {
    cvt_kernel<<<256, 256, 0, stream>>>((const float*)d_in[IW[g]], wih[g],
                                        3 * HH[g], DIN[g], KIN[g]);
    cvt_kernel<<<256, 256, 0, stream>>>((const float*)d_in[IW[g] + 1], whh[g],
                                        3 * HH[g], HH[g], HH[g]);
  }
  cvt_kernel<<<256, 256, 0, stream>>>((const float*)d_in[5], wmout, 1024, 1024, 1024);
  for (int m = 0; m < 10; ++m)
    cvt_kernel<<<128, 256, 0, stream>>>((const float*)d_in[ILIN[m]], wlin[m],
                                        512, 512, 512);

  init_kernel<<<32, 256, 0, stream>>>(p);
  rnet_kernel<<<NBLOCKS, NTHREADS, 0, stream>>>(p);
}
